// RotaryPEMultiHeadAttention_42777874268301
// MI455X (gfx1250) — compile-verified, matrix operand order checked by reading, not run on hardware
//
#include <hip/hip_runtime.h>
#include <hip/hip_bf16.h>

// ---------------------------------------------------------------------------
// RotaryPE MHA for MI455X (gfx1250): bf16 WMMA everywhere, flash-style attn.
// B=4, CH=512, T=2048, H=8, D=64, d_rope=32
// ---------------------------------------------------------------------------

typedef __bf16 v16bf __attribute__((ext_vector_type(16)));
typedef __bf16 v8bf  __attribute__((ext_vector_type(8)));
typedef float  v8f   __attribute__((ext_vector_type(8)));

#define WMMA_BF16(a, b, c) \
  __builtin_amdgcn_wmma_f32_16x16x32_bf16(false, (a), false, (b), (short)0, (c), false, false)

// Load one 16-bit A/B fragment for a 16x16x32 WMMA from a row-major operand
// (A rows, or rows of B^T). Per-lane: row = row0 + lane%16,
// K-chunks at k0 + (lane/16)*8 and +16.  Two contiguous 16B loads.
__device__ __forceinline__ v16bf load_frag(const __bf16* base, int row0, int ld, int k0) {
  int lane = threadIdx.x & 31;
  const __bf16* p = base + (size_t)(row0 + (lane & 15)) * (size_t)ld + k0 + ((lane >> 4) << 3);
  v8bf lo = *(const v8bf*)p;
  v8bf hi = *(const v8bf*)(p + 16);
  return __builtin_shufflevector(lo, hi, 0,1,2,3,4,5,6,7,8,9,10,11,12,13,14,15);
}

// ------------------------- convert / transpose -----------------------------

// in: f32 [B][C][T]  ->  out: bf16 [B][T][C]
__global__ void k_transpose_cvt(const float* __restrict__ in, __bf16* __restrict__ out,
                                int C, int T) {
  size_t idx = (size_t)blockIdx.x * blockDim.x + threadIdx.x;  // over B*T*C
  int c = (int)(idx % C);
  size_t bt = idx / C;
  int t = (int)(bt % T);
  int b = (int)(bt / T);
  out[idx] = (__bf16)in[((size_t)b * C + c) * T + t];
}

// plain f32 -> bf16
__global__ void k_cvt(const float* __restrict__ in, __bf16* __restrict__ out, int n) {
  int i = blockIdx.x * blockDim.x + threadIdx.x;
  if (i < n) out[i] = (__bf16)in[i];
}

// ------------------------------ GEMM (NT) ----------------------------------
// Out = A(MxK) * Bt(NxK)^T + bias[M], per-batch Bt.
// Block: 128 threads = 4 waves as 2x2; block tile 64x128; wave tile 32x64.
// Per K-step: 2 A-frags + 4 B-frags -> 8 WMMAs (A reused x4, B reused x2).
// mode 0: store bf16 to [b][h][T][64]   (Q / K head layout)
// mode 1: store bf16 to [b][h][64][T]   (V transposed, d-major)
// mode 2: store f32  to [b][M][N]       (final output)
__global__ void k_gemm_nt(const __bf16* __restrict__ A, const __bf16* __restrict__ Bt,
                          const float* __restrict__ bias, void* __restrict__ Out,
                          int M, int N, int K, int mode) {
  int lane = threadIdx.x & 31;
  int wave = threadIdx.x >> 5;  // 0..3
  int b = blockIdx.z;
  int m0 = blockIdx.y * 64 + (wave >> 1) * 32;
  int n0 = blockIdx.x * 128 + (wave & 1) * 64;
  const __bf16* Bb = Bt + (size_t)b * N * K;

  v8f acc[2][4] = {};
  for (int k = 0; k < K; k += 32) {
    v16bf a0 = load_frag(A, m0,      K, k);
    v16bf a1 = load_frag(A, m0 + 16, K, k);
    v16bf bb[4];
#pragma unroll
    for (int j = 0; j < 4; ++j) bb[j] = load_frag(Bb, n0 + j * 16, K, k);
#pragma unroll
    for (int j = 0; j < 4; ++j) {
      acc[0][j] = WMMA_BF16(a0, bb[j], acc[0][j]);
      acc[1][j] = WMMA_BF16(a1, bb[j], acc[1][j]);
    }
    // gfx1250 global_prefetch_b8: pull next K-block of A/B into cache.
    if (k + 32 < K) {
      __builtin_prefetch(A  + (size_t)(m0 + (lane & 15)) * K + k + 32, 0, 3);
      __builtin_prefetch(Bb + (size_t)(n0 + (lane & 15)) * K + k + 32, 0, 3);
    }
  }

  // C-layout: VGPR r, lanes 0-15: (M=r, N=lane); lanes 16-31: (M=r+8, N=lane-16)
  int half8 = (lane >> 4) << 3;
  int nl = lane & 15;
#pragma unroll
  for (int i = 0; i < 2; ++i)
#pragma unroll
    for (int j = 0; j < 4; ++j)
#pragma unroll
      for (int r = 0; r < 8; ++r) {
        int mm = m0 + i * 16 + r + half8;
        int nn = n0 + j * 16 + nl;
        float v = acc[i][j][r] + bias[mm];
        if (mode == 0) {            // Q/K: [b][h][T][64]
          int h = mm >> 6, d = mm & 63;
          ((__bf16*)Out)[(((size_t)b * 8 + h) * (size_t)N + nn) * 64 + d] = (__bf16)v;
        } else if (mode == 1) {     // Vt: [b][h][64][T]
          int h = mm >> 6, d = mm & 63;
          ((__bf16*)Out)[(((size_t)b * 8 + h) * 64 + d) * (size_t)N + nn] = (__bf16)v;
        } else {                    // f32: [b][M][N]
          ((float*)Out)[((size_t)b * M + mm) * (size_t)N + nn] = v;
        }
      }
}

// -------------------------------- RoPE -------------------------------------
// X: bf16 [B*H][T][64]; rotate dims (j, j+16) for j<16 by angle t*theta_j.
__global__ void k_rope(__bf16* __restrict__ X) {
  int idx = blockIdx.x * blockDim.x + threadIdx.x;  // (bh, t, j)
  int j = idx & 15;
  int t = (idx >> 4) & 2047;
  int bh = idx >> 15;
  __bf16* p = X + ((size_t)bh * 2048 + t) * 64;
  float theta = __powf(10000.f, -(float)j * (1.f / 16.f));
  float ang = (float)t * theta;
  float s, c;
  __sincosf(ang, &s, &c);
  float x0 = (float)p[j], x1 = (float)p[j + 16];
  p[j]      = (__bf16)(x0 * c - x1 * s);
  p[j + 16] = (__bf16)(x1 * c + x0 * s);
}

// ---------------------------- flash attention ------------------------------
// Q,K: bf16 [b][h][T][64]; Vt: bf16 [b][h][64][T]; AO: bf16 [b][T][512]
// One wave owns 32 query rows (2 q-tiles). Scores computed transposed
// (St = K * Q^T, M=k N=q) so softmax stats are per-lane (q = lane%16) and the
// exp'd probs re-pack in-lane into the PV A-fragment. K and V fragments are
// reused across both q-tiles: 16 WMMAs per 8 fragment loads per 32-key block.
__global__ void k_attn(const __bf16* __restrict__ Q, const __bf16* __restrict__ K,
                       const __bf16* __restrict__ Vt, __bf16* __restrict__ AO) {
  const int T = 2048, D = 64, H = 8;
  int lane = threadIdx.x & 31;
  int wave = threadIdx.x >> 5;  // 0..7
  int h = blockIdx.y, b = blockIdx.z;
  int q0 = blockIdx.x * 256 + wave * 32;

  const __bf16* Qb = Q  + (size_t)(b * H + h) * T * D;
  const __bf16* Kb = K  + (size_t)(b * H + h) * T * D;
  const __bf16* Vb = Vt + (size_t)(b * H + h) * D * T;

  // Q as B-operand (rows = q, K-dim = d), cached for the whole pass.
  v16bf qb[2][2];
#pragma unroll
  for (int qt = 0; qt < 2; ++qt) {
    qb[qt][0] = load_frag(Qb, q0 + qt * 16, D, 0);
    qb[qt][1] = load_frag(Qb, q0 + qt * 16, D, 32);
  }

  v8f o[2][4] = {};              // O tiles: [q-tile][d-tile], C-layout
  float m[2] = {-1e30f, -1e30f};
  float l[2] = {0.f, 0.f};
  const float scale = 0.125f;    // 1/sqrt(64)
  int half8 = (lane >> 4) << 3;

  for (int ks = 0; ks < T; ks += 32) {
    // K as A-operand: [k-subtile][d-half]; shared by both q-tiles.
    v16bf ka00 = load_frag(Kb, ks,      D, 0);
    v16bf ka01 = load_frag(Kb, ks,      D, 32);
    v16bf ka10 = load_frag(Kb, ks + 16, D, 0);
    v16bf ka11 = load_frag(Kb, ks + 16, D, 32);

    v16bf pa[2];
#pragma unroll
    for (int qt = 0; qt < 2; ++qt) {
      v8f s0 = {}, s1 = {};
      s0 = WMMA_BF16(ka00, qb[qt][0], s0);
      s0 = WMMA_BF16(ka01, qb[qt][1], s0);
      s1 = WMMA_BF16(ka10, qb[qt][0], s1);
      s1 = WMMA_BF16(ka11, qb[qt][1], s1);

      // per-lane: q = lane%16, k_rel = r + 8*(lane/16) (+16 for s1)
      float p0[8], p1[8];
      float mx = -1e30f;
#pragma unroll
      for (int r = 0; r < 8; ++r) {
        p0[r] = s0[r] * scale;
        p1[r] = s1[r] * scale;
        mx = fmaxf(mx, fmaxf(p0[r], p1[r]));
      }
      mx = fmaxf(mx, __shfl_xor(mx, 16, 32));
      float mn = fmaxf(m[qt], mx);
      float alpha = __expf(m[qt] - mn);
      m[qt] = mn;

      float sum = 0.f;
#pragma unroll
      for (int r = 0; r < 8; ++r) {
        p0[r] = __expf(p0[r] - mn);
        p1[r] = __expf(p1[r] - mn);
        sum += p0[r] + p1[r];
      }
      sum += __shfl_xor(sum, 16, 32);
      l[qt] = l[qt] * alpha + sum;

      // Transposed-C layout == A-fragment layout: pack in-lane.
#pragma unroll
      for (int r = 0; r < 8; ++r) {
        pa[qt][r]     = (__bf16)p0[r];
        pa[qt][r + 8] = (__bf16)p1[r];
      }

      // Rescale O: alpha lives at lane q; O row q = r + 8*(lane/16).
      float bc[8];
#pragma unroll
      for (int r = 0; r < 8; ++r) bc[r] = __shfl(alpha, r + half8, 32);
#pragma unroll
      for (int t = 0; t < 4; ++t)
#pragma unroll
        for (int r = 0; r < 8; ++r) o[qt][t][r] *= bc[r];
    }

    // O += P * V : B-operand rows = Vt rows (d-major, contiguous in k).
#pragma unroll
    for (int t = 0; t < 4; ++t) {
      v16bf vb = load_frag(Vb, t * 16, T, ks);
      o[0][t] = WMMA_BF16(pa[0], vb, o[0][t]);
      o[1][t] = WMMA_BF16(pa[1], vb, o[1][t]);
    }
  }

  // Normalize and scatter to AO [b][t][h*64 + d] (row-major (t, c) for final NT GEMM).
  int nl = lane & 15;
#pragma unroll
  for (int qt = 0; qt < 2; ++qt) {
    float bl[8];
#pragma unroll
    for (int r = 0; r < 8; ++r) bl[r] = __shfl(l[qt], r + half8, 32);
#pragma unroll
    for (int t = 0; t < 4; ++t)
#pragma unroll
      for (int r = 0; r < 8; ++r) {
        int tq = q0 + qt * 16 + r + half8;
        int col = h * 64 + t * 16 + nl;
        AO[((size_t)b * T + tq) * 512 + col] = (__bf16)(o[qt][t][r] / bl[r]);
      }
  }
}

// ------------------------------ launcher -----------------------------------

extern "C" void kernel_launch(void* const* d_in, const int* in_sizes, int n_in,
                              void* d_out, int out_size, void* d_ws, size_t ws_size,
                              hipStream_t stream) {
  const int B = 4, CH = 512, T = 2048, H = 8;
  const float* x  = (const float*)d_in[0];
  const float* c  = (const float*)d_in[1];
  const float* Wq = (const float*)d_in[2];
  const float* bq = (const float*)d_in[3];
  const float* Wk = (const float*)d_in[4];
  const float* bk = (const float*)d_in[5];
  const float* Wv = (const float*)d_in[6];
  const float* bv = (const float*)d_in[7];
  const float* Wo = (const float*)d_in[8];
  const float* bo = (const float*)d_in[9];

  char* ws = (char*)d_ws;
  size_t off = 0;
  const size_t ACT = (size_t)B * T * CH * sizeof(__bf16);   // 8 MiB
  const size_t WSZ = (size_t)CH * CH * sizeof(__bf16);      // 512 KiB
  __bf16* Xt   = (__bf16*)(ws + off); off += ACT;
  __bf16* Ct   = (__bf16*)(ws + off); off += ACT;
  __bf16* Wq16 = (__bf16*)(ws + off); off += WSZ;
  __bf16* Wk16 = (__bf16*)(ws + off); off += WSZ;
  __bf16* Wv16 = (__bf16*)(ws + off); off += WSZ;
  __bf16* Wo16 = (__bf16*)(ws + off); off += WSZ;
  __bf16* Qw   = (__bf16*)(ws + off); off += ACT;
  __bf16* Kw   = (__bf16*)(ws + off); off += ACT;
  __bf16* Vtw  = (__bf16*)(ws + off); off += ACT;
  __bf16* AO   = (__bf16*)(ws + off); off += ACT;

  // 1) activations: f32 [B][C][T] -> bf16 [B][T][C]
  {
    size_t n = (size_t)B * T * CH;
    k_transpose_cvt<<<dim3((unsigned)(n / 256)), 256, 0, stream>>>(x, Xt, CH, T);
    k_transpose_cvt<<<dim3((unsigned)(n / 256)), 256, 0, stream>>>(c, Ct, CH, T);
  }
  // 2) weights -> bf16
  {
    int n = CH * CH;
    k_cvt<<<dim3(n / 256), 256, 0, stream>>>(Wq, Wq16, n);
    k_cvt<<<dim3(n / 256), 256, 0, stream>>>(Wk, Wk16, n);
    k_cvt<<<dim3(n / 256), 256, 0, stream>>>(Wv, Wv16, n);
    k_cvt<<<dim3(n / 256), 256, 0, stream>>>(Wo, Wo16, n);
  }
  // 3) projections (M=512, N=2048, K=512 per batch)
  {
    dim3 grid(T / 128, CH / 64, B);
    k_gemm_nt<<<grid, 128, 0, stream>>>(Wq16, Xt, bq, Qw,  CH, T, CH, 0);
    k_gemm_nt<<<grid, 128, 0, stream>>>(Wk16, Ct, bk, Kw,  CH, T, CH, 0);
    k_gemm_nt<<<grid, 128, 0, stream>>>(Wv16, Ct, bv, Vtw, CH, T, CH, 1);
  }
  // 4) RoPE on Q and K
  {
    int n = B * H * T * 16;
    k_rope<<<dim3(n / 256), 256, 0, stream>>>(Qw);
    k_rope<<<dim3(n / 256), 256, 0, stream>>>(Kw);
  }
  // 5) attention (each wave: 32 q rows; block: 8 waves = 256 q rows)
  {
    dim3 grid(T / 256, H, B);
    k_attn<<<grid, 256, 0, stream>>>(Qw, Kw, Vtw, AO);
  }
  // 6) output projection -> f32 d_out [B][512][2048]
  {
    dim3 grid(T / 128, CH / 64, B);
    k_gemm_nt<<<grid, 128, 0, stream>>>(Wo16, AO, bo, d_out, CH, T, CH, 2);
  }
  (void)in_sizes; (void)n_in; (void)out_size; (void)ws_size;
}
